// LinearAttention_14422500179995
// MI455X (gfx1250) — matrix operand order repaired and checked by reading, not compiled, not run on hardware
//
#include <hip/hip_runtime.h>

// Problem dims (fixed by the reference)
#define N_ROWS 8192
#define XDIM   1024
#define HDIM   1024
#define YDIM   1024

typedef __attribute__((ext_vector_type(16))) __bf16 v16bf;
typedef __attribute__((ext_vector_type(8)))  __bf16 v8bf;
typedef __attribute__((ext_vector_type(8)))  float  v8f;

// ---------------- workspace layout (bytes), total ~55.1 MB ----------------
#define XB_OFF   ((size_t)0)                              // x  bf16 (N,X):  16 MB
#define WKT_OFF  ((size_t)16 * 1024 * 1024)               // Wk^T bf16 (X,H): 2 MB
#define WQT_OFF  (WKT_OFF + (size_t)2 * 1024 * 1024)      // Wq^T bf16 (X,H): 2 MB
#define GT_OFF   (WQT_OFF + (size_t)2 * 1024 * 1024)      // Gt bf16 (X,X):   2 MB
#define T_OFF    (GT_OFF + (size_t)2 * 1024 * 1024)       // T' f32 (N,X):   32 MB
#define CS_OFF   (T_OFF + (size_t)32 * 1024 * 1024)       // chunk sums:      1 MB
#define R_OFF    (CS_OFF + (size_t)1 * 1024 * 1024)       // r f32:          32 KB
#define U_OFF    (R_OFF + (size_t)32 * 1024)              // u f32:           4 KB
#define C_OFF    (U_OFF + (size_t)4 * 1024)               // c f32 scalar
#define W_OFF    (C_OFF + (size_t)256)                    // w f32:           4 KB

#define CHUNKS 256
#define CROWS  (N_ROWS / CHUNKS)   // 32 rows per chunk

// ---------------- f32 -> bf16 conversion (8 elems/thread) ----------------
__global__ __launch_bounds__(256) void cvt_f32_bf16(const float* __restrict__ in,
                                                    __bf16* __restrict__ out, int n8) {
  int idx = blockIdx.x * blockDim.x + threadIdx.x;
  if (idx >= n8) return;
  const float4* p = (const float4*)in + (size_t)idx * 2;
  float4 a = p[0], b = p[1];
  v8bf o;
  o[0] = (__bf16)a.x; o[1] = (__bf16)a.y; o[2] = (__bf16)a.z; o[3] = (__bf16)a.w;
  o[4] = (__bf16)b.x; o[5] = (__bf16)b.y; o[6] = (__bf16)b.z; o[7] = (__bf16)b.w;
  *((v8bf*)out + idx) = o;
}

// ---------------- LDS-tiled transpose + convert: (H,X) f32 -> (X,H) bf16 ----------------
__global__ __launch_bounds__(256) void transpose_cvt(const float* __restrict__ in,
                                                     __bf16* __restrict__ out) {
  __shared__ float tile[32][33];
  const int bx = blockIdx.x * 32;          // x base
  const int by = blockIdx.y * 32;          // h base
  const int tx = threadIdx.x & 31;
  const int ty = threadIdx.x >> 5;         // 0..7
#pragma unroll
  for (int rr = 0; rr < 4; ++rr)
    tile[ty + 8 * rr][tx] = in[(size_t)(by + ty + 8 * rr) * XDIM + bx + tx];
  __syncthreads();
#pragma unroll
  for (int rr = 0; rr < 4; ++rr)
    out[(size_t)(bx + ty + 8 * rr) * HDIM + by + tx] = (__bf16)tile[tx][ty + 8 * rr];
}

// ---------------- u = Wq^T bk, w = Wk^T bq (f32, coalesced column sweeps) ----------------
__global__ __launch_bounds__(256) void make_uw(const float* __restrict__ Wq,
                                               const float* __restrict__ bk,
                                               const float* __restrict__ Wk,
                                               const float* __restrict__ bq,
                                               float* __restrict__ u,
                                               float* __restrict__ w) {
  const int col = blockIdx.x * 256 + threadIdx.x;  // 0..2047
  if (col < XDIM) {
    float s = 0.f;
    for (int h = 0; h < HDIM; ++h) s += Wq[(size_t)h * XDIM + col] * bk[h];
    u[col] = s;
  } else {
    const int c2 = col - XDIM;
    float s = 0.f;
    for (int h = 0; h < HDIM; ++h) s += Wk[(size_t)h * XDIM + c2] * bq[h];
    w[c2] = s;
  }
}

// ---------------- c = bq . bk (single block reduce) ----------------
__global__ __launch_bounds__(256) void make_c(const float* __restrict__ bq,
                                              const float* __restrict__ bk,
                                              float* __restrict__ cout) {
  __shared__ float red[8];
  const int t = threadIdx.x;
  float s = 0.f;
  for (int h = t; h < HDIM; h += 256) s += bq[h] * bk[h];
#pragma unroll
  for (int off = 16; off; off >>= 1) s += __shfl_xor(s, off, 32);
  if ((t & 31) == 0) red[t >> 5] = s;
  __syncthreads();
  if (t == 0) {
    float tot = 0.f;
#pragma unroll
    for (int wv = 0; wv < 8; ++wv) tot += red[wv];
    *cout = tot;
  }
}

// ---------------- small WMMA GEMM: Gt[x2,x1] = sum_h wkT[x2,h] * wqT[x1,h], out bf16 ----------------
// block tile 128(M) x 64(N), wave tile 32x32 (2x2 WMMA)
__global__ __launch_bounds__(256) void gemm_G(const __bf16* __restrict__ wkT,
                                              const __bf16* __restrict__ wqT,
                                              __bf16* __restrict__ gt) {
  const int lane = threadIdx.x & 31;
  const int wave = threadIdx.x >> 5;
  const int r0 = blockIdx.y * 128 + (wave >> 1) * 32;
  const int c0 = blockIdx.x * 64 + (wave & 1) * 32;
  const int hl = lane >> 4, l15 = lane & 15;
  union Frag { v16bf v; v8bf h[2]; };
  v8f acc[2][2] = {};
  for (int kk = 0; kk < HDIM; kk += 32) {
    Frag a[2], b[2];
#pragma unroll
    for (int i = 0; i < 2; ++i) {
      const __bf16* pa = wkT + (size_t)(r0 + i * 16 + l15) * HDIM + kk + hl * 16;
      a[i].h[0] = *(const v8bf*)pa;
      a[i].h[1] = *(const v8bf*)(pa + 8);
      const __bf16* pb = wqT + (size_t)(c0 + i * 16 + l15) * HDIM + kk + hl * 16;
      b[i].v = *(const v16bf*)pb;
    }
#pragma unroll
    for (int i = 0; i < 2; ++i)
#pragma unroll
      for (int j = 0; j < 2; ++j)
        acc[i][j] = __builtin_amdgcn_wmma_f32_16x16x32_bf16(
            false, a[i].v, false, b[j].v, (short)0, acc[i][j], false, false);
  }
#pragma unroll
  for (int i = 0; i < 2; ++i)
#pragma unroll
    for (int j = 0; j < 2; ++j) {
      const int col = c0 + j * 16 + l15;
#pragma unroll
      for (int g = 0; g < 8; ++g) {
        const int row = r0 + i * 16 + hl * 8 + g;
        gt[(size_t)row * XDIM + col] = (__bf16)acc[i][j][g];
      }
    }
}

// ---------------- big WMMA GEMM: T'[i,x2] = sum_x1 xb[i,x1]*Gt[x2,x1] + w[x2] ----------------
// block tile 128(M) x 128(N), wave tile 32x64 (2x4 WMMA)
__global__ __launch_bounds__(256) void gemm_T(const __bf16* __restrict__ xb,
                                              const __bf16* __restrict__ gt,
                                              const float* __restrict__ w,
                                              float* __restrict__ tout) {
  const int lane = threadIdx.x & 31;
  const int wave = threadIdx.x >> 5;
  const int r0 = blockIdx.y * 128 + (wave >> 1) * 32;   // M rows
  const int c0 = blockIdx.x * 128 + (wave & 1) * 64;    // N cols (x2)
  const int hl = lane >> 4, l15 = lane & 15;
  union Frag { v16bf v; v8bf h[2]; };
  v8f acc[2][4] = {};
  for (int kk = 0; kk < XDIM; kk += 32) {
    Frag a[2], b[4];
#pragma unroll
    for (int i = 0; i < 2; ++i) {
      const __bf16* pa = xb + (size_t)(r0 + i * 16 + l15) * XDIM + kk + hl * 16;
      a[i].h[0] = *(const v8bf*)pa;
      a[i].h[1] = *(const v8bf*)(pa + 8);
      if (i == 0) __builtin_prefetch(pa + 32, 0, 3);   // next K-slab -> global_prefetch
    }
#pragma unroll
    for (int j = 0; j < 4; ++j) {
      const __bf16* pb = gt + (size_t)(c0 + j * 16 + l15) * XDIM + kk + hl * 16;
      b[j].v = *(const v16bf*)pb;
    }
#pragma unroll
    for (int i = 0; i < 2; ++i)
#pragma unroll
      for (int j = 0; j < 4; ++j)
        acc[i][j] = __builtin_amdgcn_wmma_f32_16x16x32_bf16(
            false, a[i].v, false, b[j].v, (short)0, acc[i][j], false, false);
  }
#pragma unroll
  for (int i = 0; i < 2; ++i)
#pragma unroll
    for (int j = 0; j < 4; ++j) {
      const int col = c0 + j * 16 + l15;
      const float wb = w[col];
#pragma unroll
      for (int g = 0; g < 8; ++g) {
        const int row = r0 + i * 16 + hl * 8 + g;
        tout[(size_t)row * XDIM + col] = acc[i][j][g] + wb;
      }
    }
}

// ---------------- chunk sums of x rows ----------------
__global__ __launch_bounds__(256) void chunk_sum(const float* __restrict__ x,
                                                 float* __restrict__ cs) {
  const int c = blockIdx.x;
  const int h = threadIdx.x * 4;
  const float* p = x + (size_t)c * CROWS * XDIM + h;
  float4 s = {0.f, 0.f, 0.f, 0.f};
  for (int i = 0; i < CROWS; ++i) {
    float4 v = *(const float4*)(p + (size_t)i * XDIM);
    s.x += v.x; s.y += v.y; s.z += v.z; s.w += v.w;
  }
  *(float4*)(cs + (size_t)c * XDIM + h) = s;
}

// ---------------- exclusive prefix over chunk sums (per column, in place) ----------------
__global__ __launch_bounds__(256) void chunk_scan(float* __restrict__ cs) {
  const int h = blockIdx.x * blockDim.x + threadIdx.x;
  float run = 0.f;
  for (int c = 0; c < CHUNKS; ++c) {
    float t = cs[(size_t)c * XDIM + h];
    cs[(size_t)c * XDIM + h] = run;
    run += t;
  }
}

// ---------------- per-chunk scan: r_i = T'_i . S_i + i*(x_i . u + c) ----------------
__global__ __launch_bounds__(256) void scan_dot(const float* __restrict__ T,
                                                const float* __restrict__ x,
                                                const float* __restrict__ cs,
                                                const float* __restrict__ u,
                                                const float* __restrict__ cscal,
                                                float* __restrict__ r) {
  __shared__ float red1[8], red2[8];
  const int c = blockIdx.x;
  const int t = threadIdx.x;
  const int h = t * 4;
  const float cc = *cscal;
  float4 S  = *(const float4*)(cs + (size_t)c * XDIM + h);  // exclusive prefix of x rows
  float4 uv = *(const float4*)(u + h);
  const size_t base = (size_t)c * CROWS * XDIM + h;
  for (int i = 0; i < CROWS; ++i) {
    float4 tv = *(const float4*)(T + base + (size_t)i * XDIM);
    float4 xv = *(const float4*)(x + base + (size_t)i * XDIM);
    float p1 = tv.x * S.x + tv.y * S.y + tv.z * S.z + tv.w * S.w;
    float p2 = xv.x * uv.x + xv.y * uv.y + xv.z * uv.z + xv.w * uv.w;
#pragma unroll
    for (int off = 16; off; off >>= 1) {
      p1 += __shfl_xor(p1, off, 32);
      p2 += __shfl_xor(p2, off, 32);
    }
    if ((t & 31) == 0) { red1[t >> 5] = p1; red2[t >> 5] = p2; }
    __syncthreads();
    if (t == 0) {
      float t1 = 0.f, t2 = 0.f;
#pragma unroll
      for (int wv = 0; wv < 8; ++wv) { t1 += red1[wv]; t2 += red2[wv]; }
      const int gi = c * CROWS + i;
      r[gi] = t1 + (float)gi * (t2 + cc);
    }
    S.x += xv.x; S.y += xv.y; S.z += xv.z; S.w += xv.w;
    __syncthreads();
  }
}

// ---------------- broadcast r across all Y output columns ----------------
__global__ __launch_bounds__(256) void broadcast_r(const float* __restrict__ r,
                                                   float* __restrict__ out) {
  const size_t idx = (size_t)blockIdx.x * blockDim.x + threadIdx.x;  // one float4 each
  const int row = (int)(idx >> 8);                                   // Y/4 = 256 vec4/row
  const float v = r[row];
  float4 o = {v, v, v, v};
  *((float4*)out + idx) = o;
}

extern "C" void kernel_launch(void* const* d_in, const int* in_sizes, int n_in,
                              void* d_out, int out_size, void* d_ws, size_t ws_size,
                              hipStream_t stream) {
  (void)in_sizes; (void)n_in; (void)out_size; (void)ws_size;
  const float* x  = (const float*)d_in[0];
  const float* Wk = (const float*)d_in[1];
  const float* bk = (const float*)d_in[2];
  // d_in[3] = Wv, d_in[4] = bv : dead (reference overwrites v with ones)
  const float* Wq = (const float*)d_in[5];
  const float* bq = (const float*)d_in[6];
  float* out = (float*)d_out;

  char* ws = (char*)d_ws;
  __bf16* xb  = (__bf16*)(ws + XB_OFF);
  __bf16* wkT = (__bf16*)(ws + WKT_OFF);
  __bf16* wqT = (__bf16*)(ws + WQT_OFF);
  __bf16* gt  = (__bf16*)(ws + GT_OFF);
  float*  tb  = (float*)(ws + T_OFF);
  float*  cs  = (float*)(ws + CS_OFF);
  float*  rb  = (float*)(ws + R_OFF);
  float*  ub  = (float*)(ws + U_OFF);
  float*  cb  = (float*)(ws + C_OFF);
  float*  wb  = (float*)(ws + W_OFF);

  // 1) precision prep: x -> bf16; Wk,Wq -> transposed bf16 (K-contiguous operands)
  cvt_f32_bf16<<<(N_ROWS * XDIM / 8) / 256, 256, 0, stream>>>(x, xb, N_ROWS * XDIM / 8);
  transpose_cvt<<<dim3(32, 32), 256, 0, stream>>>(Wk, wkT);
  transpose_cvt<<<dim3(32, 32), 256, 0, stream>>>(Wq, wqT);

  // 2) bias-derived vectors/scalar in f32: u = Wq^T bk, w = Wk^T bq, c = bq.bk
  make_uw<<<8, 256, 0, stream>>>(Wq, bk, Wk, bq, ub, wb);
  make_c<<<1, 256, 0, stream>>>(bq, bk, cb);

  // 3) Gt = Wk^T Wq  (so that T = x @ G uses K-contiguous B fragments)
  gemm_G<<<dim3(XDIM / 64, XDIM / 128), 256, 0, stream>>>(wkT, wqT, gt);

  // 4) T' = x @ G + w  (big WMMA GEMM, w folded as epilogue bias)
  gemm_T<<<dim3(XDIM / 128, N_ROWS / 128), 256, 0, stream>>>(xb, gt, wb, tb);

  // 5) blocked exclusive prefix scan of raw x rows fused with the two dots
  chunk_sum<<<CHUNKS, 256, 0, stream>>>(x, cs);
  chunk_scan<<<XDIM / 256, 256, 0, stream>>>(cs);
  scan_dot<<<CHUNKS, 256, 0, stream>>>(tb, x, cs, ub, cb, rb);

  // 6) out[i, :] = r[i]
  broadcast_r<<<(N_ROWS * YDIM / 4) / 256, 256, 0, stream>>>(rb, out);
}
